// GraphSAGE_52381421142394
// MI455X (gfx1250) — compile-verified, matrix operand order checked by reading, not run on hardware
//
#include <hip/hip_runtime.h>
#include <hip/hip_bf16.h>
#include <math.h>

// ---------------- problem constants (match reference) ----------------
#define B_GRAPHS 4
#define NPG      4096
#define N_NODES  (B_GRAPHS * NPG)      // 16384
#define E_EDGES  131072
#define DIN      256
#define DHID     512
#define DOUT     512
#define KEEP1    (NPG / 2)             // 2048 per graph
#define KEEP2    (NPG / 4)             // 1024 per graph

typedef __attribute__((ext_vector_type(16))) __bf16 v16bf;
typedef __attribute__((ext_vector_type(8)))  float  v8f;

union FragBF { v16bf v; unsigned int u[8]; uint4 q[2]; };

// round-to-nearest-even f32 -> bf16, packed pair into one dword
static __device__ __forceinline__ unsigned int f2bf_pk(float lo, float hi) {
    unsigned int ul = __float_as_uint(lo);
    unsigned int uh = __float_as_uint(hi);
    unsigned int bl = (ul + 0x7FFFu + ((ul >> 16) & 1u)) >> 16;
    unsigned int bh = (uh + 0x7FFFu + ((uh >> 16) & 1u)) >> 16;
    return (bl & 0xFFFFu) | (bh << 16);
}

// ---------------- generic fills ----------------
__global__ void fill_f32(float* __restrict__ p, float v, long long n) {
    long long i = (long long)blockIdx.x * blockDim.x + threadIdx.x;
    if (i < n) p[i] = v;
}
__global__ void fill_i32(int* __restrict__ p, int v, long long n) {
    long long i = (long long)blockIdx.x * blockDim.x + threadIdx.x;
    if (i < n) p[i] = v;
}

// ---------------- bf16 packing ----------------
// row-major f32 [R x C] -> packed bf16 pairs, one dword per 2 elements
__global__ void pack_bf16_rows(const float* __restrict__ in, unsigned int* __restrict__ out,
                               long long n2) {
    long long i = (long long)blockIdx.x * blockDim.x + threadIdx.x;
    if (i >= n2) return;
    const float2 v = ((const float2*)in)[i];
    out[i] = f2bf_pk(v.x, v.y);
}
// W [K x Nout] row-major f32 -> Wt packed [Nout x K/2] dwords (pairs along K)
__global__ void pack_w_transpose(const float* __restrict__ W, unsigned int* __restrict__ out,
                                 int K, int Nout) {
    int tid = blockIdx.x * blockDim.x + threadIdx.x;
    int kd = K >> 1;
    if (tid >= Nout * kd) return;
    int n = tid / kd, k2 = tid % kd;
    out[tid] = f2bf_pk(W[(size_t)(2 * k2) * Nout + n], W[(size_t)(2 * k2 + 1) * Nout + n]);
}

// ---------------- graph scatter (segment_sum) ----------------
__global__ void scatter_add_feats(const float* __restrict__ x, const int* __restrict__ src,
                                  const int* __restrict__ dst, const float* __restrict__ mask,
                                  float* __restrict__ agg, int E, int D) {
    long long tid = (long long)blockIdx.x * blockDim.x + threadIdx.x;
    int chunks = D >> 2;
    long long total = (long long)E * chunks;
    if (tid >= total) return;
    int e = (int)(tid / chunks);
    int c = (int)(tid % chunks);
    float m = mask[e];
    if (m == 0.0f) return;
    const float4 v = ((const float4*)(x + (size_t)src[e] * D))[c];
    float* base = agg + (size_t)dst[e] * D + (size_t)c * 4;
    atomicAdd(base + 0, v.x * m);
    atomicAdd(base + 1, v.y * m);
    atomicAdd(base + 2, v.z * m);
    atomicAdd(base + 3, v.w * m);
}

__global__ void scatter_count(const int* __restrict__ dst, const float* __restrict__ mask,
                              float* __restrict__ cnt, int E) {
    int e = blockIdx.x * blockDim.x + threadIdx.x;
    if (e >= E) return;
    float m = mask[e];
    if (m != 0.0f) atomicAdd(&cnt[dst[e]], m);
}

// mean = agg / max(cnt,1), emitted directly as packed bf16 pairs (GEMM A operand)
__global__ void mean_div_pack(const float* __restrict__ agg, const float* __restrict__ cnt,
                              unsigned int* __restrict__ outpk, long long n, int D) {
    long long i = (long long)blockIdx.x * blockDim.x + threadIdx.x;
    int d2 = D >> 1;
    long long total = n * (long long)d2;
    if (i >= total) return;
    long long row = i / d2;
    float inv = 1.0f / fmaxf(cnt[row], 1.0f);
    const float2 v = ((const float2*)agg)[i];
    outpk[i] = f2bf_pk(v.x * inv, v.y * inv);
}

// ---------------- fused dual-GEMM: out = relu(A1@W1 + A2@W2 + bias) ----------------
// A1/A2: packed bf16 [M x K/2] dwords. W1t/W2t: packed bf16 transposed [Nout x K/2] dwords.
// One wave computes a 16x64 output tile (4 accumulators); A fragments reused across N-tiles.
// Fragment dword index (lane half h, vgpr v): k0/2 + (v>>2)*8 + h*4 + (v&3)
//   -> u[0..3] and u[4..7] are each 4 consecutive dwords: two global_load_b128 per fragment.
__global__ __launch_bounds__(32)
void gemm_dual_bias_relu(const unsigned int* __restrict__ A1, const unsigned int* __restrict__ A2,
                         const unsigned int* __restrict__ W1t, const unsigned int* __restrict__ W2t,
                         const float* __restrict__ bias, float* __restrict__ out,
                         int K, int Nout) {
    const int lane = threadIdx.x;
    const int l15  = lane & 15;
    const int half = lane >> 4;
    const int row0 = blockIdx.x * 16;
    const int col0 = blockIdx.y * 64;
    const int kd   = K >> 1;                         // dwords per row

    const unsigned int* pa1 = A1 + (size_t)(row0 + l15) * kd + half * 4;
    const unsigned int* pa2 = A2 + (size_t)(row0 + l15) * kd + half * 4;

    v8f acc[4] = {{}, {}, {}, {}};
    for (int k0d = 0; k0d < kd; k0d += 16) {
        if (k0d + 16 < kd) {   // warm next K-slab of the streamed (big) operands
            __builtin_prefetch(pa1 + k0d + 16, 0, 0);
            __builtin_prefetch(pa2 + k0d + 16, 0, 0);
        }
        FragBF a1, a2;
        a1.q[0] = *(const uint4*)(pa1 + k0d);
        a1.q[1] = *(const uint4*)(pa1 + k0d + 8);
        a2.q[0] = *(const uint4*)(pa2 + k0d);
        a2.q[1] = *(const uint4*)(pa2 + k0d + 8);
#pragma unroll
        for (int t = 0; t < 4; ++t) {
            const unsigned int* pw1 = W1t + (size_t)(col0 + 16 * t + l15) * kd + half * 4 + k0d;
            const unsigned int* pw2 = W2t + (size_t)(col0 + 16 * t + l15) * kd + half * 4 + k0d;
            FragBF b1, b2;
            b1.q[0] = *(const uint4*)(pw1);
            b1.q[1] = *(const uint4*)(pw1 + 8);
            b2.q[0] = *(const uint4*)(pw2);
            b2.q[1] = *(const uint4*)(pw2 + 8);
            acc[t] = __builtin_amdgcn_wmma_f32_16x16x32_bf16(false, a1.v, false, b1.v,
                                                            (short)0, acc[t], false, false);
            acc[t] = __builtin_amdgcn_wmma_f32_16x16x32_bf16(false, a2.v, false, b2.v,
                                                            (short)0, acc[t], false, false);
        }
    }
    const int mOut = row0 + half * 8;   // D layout: lanes 0-15 -> M=r, lanes 16-31 -> M=8+r
#pragma unroll
    for (int t = 0; t < 4; ++t) {
        const int nB = col0 + 16 * t + l15;
        const float bv = bias[nB];
#pragma unroll
        for (int r = 0; r < 8; ++r) {
            float v = acc[t][r] + bv;
            out[(size_t)(mOut + r) * Nout + nB] = fmaxf(v, 0.0f);
        }
    }
}

// ---------------- pooling score: tanh(agg.Wrel + b + h.Wroot) ----------------
__global__ __launch_bounds__(128)
void score_tanh(const float* __restrict__ agg, const float* __restrict__ h,
                const float* __restrict__ Wrel, const float* __restrict__ brel,
                const float* __restrict__ Wroot, float* __restrict__ score, int D) {
    __shared__ float red[128];
    const int i = blockIdx.x, t = threadIdx.x;
    float s = 0.0f;
    for (int d = t; d < D; d += 128)
        s += agg[(size_t)i * D + d] * Wrel[d] + h[(size_t)i * D + d] * Wroot[d];
    red[t] = s;
    __syncthreads();
    for (int off = 64; off > 0; off >>= 1) {
        if (t < off) red[t] += red[t + off];
        __syncthreads();
    }
    if (t == 0) score[i] = tanhf(red[0] + brel[0]);
}

// ---------------- sorted per-graph top-k via LDS bitonic sort ----------------
// descending by score, tie -> smaller original index first (matches lax.top_k)
__global__ __launch_bounds__(1024)
void topk_sort(const float* __restrict__ score, int npg, int k,
               int* __restrict__ perm, float* __restrict__ vals) {
    __shared__ float ss[4096];
    __shared__ int   si[4096];
    const int g = blockIdx.x;
    const int t = threadIdx.x;
    const int NT = blockDim.x;
    for (int i = t; i < npg; i += NT) { ss[i] = score[g * npg + i]; si[i] = i; }
    __syncthreads();
    for (int kk = 2; kk <= npg; kk <<= 1) {
        for (int j = kk >> 1; j > 0; j >>= 1) {
            for (int i = t; i < npg; i += NT) {
                const int ixj = i ^ j;
                if (ixj > i) {
                    float s0 = ss[i], s1 = ss[ixj];
                    int   i0 = si[i], i1 = si[ixj];
                    bool before10 = (s1 > s0) || (s1 == s0 && i1 < i0);
                    bool before01 = (s0 > s1) || (s0 == s1 && i0 < i1);
                    bool descBlk  = ((i & kk) == 0);
                    bool doSwap   = descBlk ? before10 : before01;
                    if (doSwap) { ss[i] = s1; ss[ixj] = s0; si[i] = i1; si[ixj] = i0; }
                }
            }
            __syncthreads();
        }
    }
    for (int i = t; i < k; i += NT) {
        perm[g * k + i] = g * npg + si[i];
        vals[g * k + i] = ss[i];
    }
}

__global__ void set_newid(const int* __restrict__ perm, int* __restrict__ new_id, int nk) {
    int i = blockIdx.x * blockDim.x + threadIdx.x;
    if (i < nk) new_id[perm[i]] = i;
}

__global__ void gather_scale(const float* __restrict__ h, const int* __restrict__ perm,
                             const float* __restrict__ vals, float* __restrict__ out,
                             int rows, int D) {
    long long tid = (long long)blockIdx.x * blockDim.x + threadIdx.x;
    int chunks = D >> 2;
    long long total = (long long)rows * chunks;
    if (tid >= total) return;
    int r = (int)(tid / chunks);
    int c = (int)(tid % chunks);
    float4 v = ((const float4*)(h + (size_t)perm[r] * D))[c];
    float m = vals[r];
    v.x *= m; v.y *= m; v.z *= m; v.w *= m;
    ((float4*)(out + (size_t)r * D))[c] = v;
}

__global__ void reindex_edges(const int* __restrict__ src, const int* __restrict__ dst,
                              const float* __restrict__ mask, const int* __restrict__ new_id,
                              int* __restrict__ src2, int* __restrict__ dst2,
                              float* __restrict__ mask2, int E) {
    int e = blockIdx.x * blockDim.x + threadIdx.x;
    if (e >= E) return;
    int ns = new_id[src[e]], nd = new_id[dst[e]];
    float m = mask[e] * ((ns >= 0) ? 1.0f : 0.0f) * ((nd >= 0) ? 1.0f : 0.0f);
    src2[e] = (ns < 0) ? 0 : ns;
    dst2[e] = (nd < 0) ? 0 : nd;
    mask2[e] = m;
}

__global__ void write_batch(float* __restrict__ out, int n, int k) {
    int i = blockIdx.x * blockDim.x + threadIdx.x;
    if (i < n) out[i] = (float)(i / k);
}

// ---------------- host orchestration ----------------
static inline long long cdiv_ll(long long a, long long b) { return (a + b - 1) / b; }

extern "C" void kernel_launch(void* const* d_in, const int* in_sizes, int n_in,
                              void* d_out, int out_size, void* d_ws, size_t ws_size,
                              hipStream_t stream) {
    (void)in_sizes; (void)n_in; (void)out_size; (void)ws_size;
    const float* x        = (const float*)d_in[0];
    const int*   ei       = (const int*)d_in[1];
    const int*   src      = ei;
    const int*   dst      = ei + E_EDGES;
    const float* W1l      = (const float*)d_in[3];
    const float* b1l      = (const float*)d_in[4];
    const float* W1r      = (const float*)d_in[5];
    const float* Wp1_rel  = (const float*)d_in[6];
    const float* bp1      = (const float*)d_in[7];
    const float* Wp1_root = (const float*)d_in[8];
    const float* W2l      = (const float*)d_in[9];
    const float* b2l      = (const float*)d_in[10];
    const float* W2r      = (const float*)d_in[11];
    const float* Wp2_rel  = (const float*)d_in[12];
    const float* bp2      = (const float*)d_in[13];
    const float* Wp2_root = (const float*)d_in[14];
    float* out = (float*)d_out;

    // ---- carve workspace ----
    char* ws = (char*)d_ws;
    size_t off = 0;
    auto carve = [&](size_t bytes) -> void* {
        void* p = ws + off;
        off += (bytes + 255) & ~(size_t)255;
        return p;
    };
    float*        bufA   = (float*)carve((size_t)N_NODES * DHID * 4);   // agg scratch
    float*        h1     = (float*)carve((size_t)N_NODES * DHID * 4);
    float*        hp1    = (float*)carve((size_t)B_GRAPHS * KEEP1 * DHID * 4);
    float*        h2     = (float*)carve((size_t)B_GRAPHS * KEEP1 * DOUT * 4);
    unsigned int* meanpk = (unsigned int*)carve((size_t)N_NODES * DHID / 2 * 4);
    unsigned int* xpk    = (unsigned int*)carve((size_t)N_NODES * DIN / 2 * 4);
    unsigned int* hp1pk  = (unsigned int*)carve((size_t)B_GRAPHS * KEEP1 * DHID / 2 * 4);
    unsigned int* w1lpk  = (unsigned int*)carve((size_t)DIN * DHID / 2 * 4);
    unsigned int* w1rpk  = (unsigned int*)carve((size_t)DIN * DHID / 2 * 4);
    unsigned int* w2lpk  = (unsigned int*)carve((size_t)DHID * DOUT / 2 * 4);
    unsigned int* w2rpk  = (unsigned int*)carve((size_t)DHID * DOUT / 2 * 4);
    float*        cnt    = (float*)carve((size_t)N_NODES * 4);
    float*        score  = (float*)carve((size_t)N_NODES * 4);
    int*          perm   = (int*)  carve((size_t)N_NODES * 4);
    float*        vals   = (float*)carve((size_t)N_NODES * 4);
    int*          new_id = (int*)  carve((size_t)N_NODES * 4);
    float*        mask1  = (float*)carve((size_t)E_EDGES * 4);
    int*          src2   = (int*)  carve((size_t)E_EDGES * 4);
    int*          dst2   = (int*)  carve((size_t)E_EDGES * 4);
    float*        mask2  = (float*)carve((size_t)E_EDGES * 4);

    const int T = 256;
#define GRID1(n) dim3((unsigned)cdiv_ll((long long)(n), T)), dim3(T)

    // ---- one-time per-launch packing (bf16 operands for WMMA) ----
    pack_bf16_rows<<<GRID1((long long)N_NODES * DIN / 2), 0, stream>>>(x, xpk, (long long)N_NODES * DIN / 2);
    pack_w_transpose<<<GRID1(DHID * (DIN / 2)), 0, stream>>>(W1l, w1lpk, DIN, DHID);
    pack_w_transpose<<<GRID1(DHID * (DIN / 2)), 0, stream>>>(W1r, w1rpk, DIN, DHID);
    pack_w_transpose<<<GRID1(DOUT * (DHID / 2)), 0, stream>>>(W2l, w2lpk, DHID, DOUT);
    pack_w_transpose<<<GRID1(DOUT * (DHID / 2)), 0, stream>>>(W2r, w2rpk, DHID, DOUT);

    // ================= conv1: h1 = relu(mean@W1l + b1l + x@W1r) =================
    fill_f32<<<GRID1((long long)N_NODES * DIN), 0, stream>>>(bufA, 0.0f, (long long)N_NODES * DIN);
    fill_f32<<<GRID1(N_NODES), 0, stream>>>(cnt, 0.0f, N_NODES);
    fill_f32<<<GRID1(E_EDGES), 0, stream>>>(mask1, 1.0f, E_EDGES);
    scatter_add_feats<<<GRID1((long long)E_EDGES * (DIN / 4)), 0, stream>>>(x, src, dst, mask1, bufA, E_EDGES, DIN);
    scatter_count<<<GRID1(E_EDGES), 0, stream>>>(dst, mask1, cnt, E_EDGES);
    mean_div_pack<<<GRID1((long long)N_NODES * DIN / 2), 0, stream>>>(bufA, cnt, meanpk, N_NODES, DIN);
    gemm_dual_bias_relu<<<dim3(N_NODES / 16, DHID / 64), dim3(32), 0, stream>>>(
        meanpk, xpk, w1lpk, w1rpk, b1l, h1, DIN, DHID);

    // ================= pool1 =================
    fill_f32<<<GRID1((long long)N_NODES * DHID), 0, stream>>>(bufA, 0.0f, (long long)N_NODES * DHID);
    scatter_add_feats<<<GRID1((long long)E_EDGES * (DHID / 4)), 0, stream>>>(h1, src, dst, mask1, bufA, E_EDGES, DHID);
    score_tanh<<<dim3(N_NODES), dim3(128), 0, stream>>>(bufA, h1, Wp1_rel, bp1, Wp1_root, score, DHID);
    topk_sort<<<dim3(B_GRAPHS), dim3(1024), 0, stream>>>(score, NPG, KEEP1, perm, vals);
    fill_i32<<<GRID1(N_NODES), 0, stream>>>(new_id, -1, N_NODES);
    set_newid<<<GRID1(B_GRAPHS * KEEP1), 0, stream>>>(perm, new_id, B_GRAPHS * KEEP1);
    gather_scale<<<GRID1((long long)B_GRAPHS * KEEP1 * (DHID / 4)), 0, stream>>>(
        h1, perm, vals, hp1, B_GRAPHS * KEEP1, DHID);
    reindex_edges<<<GRID1(E_EDGES), 0, stream>>>(src, dst, mask1, new_id, src2, dst2, mask2, E_EDGES);

    const int N2 = B_GRAPHS * KEEP1;   // 8192 nodes after pool1
    pack_bf16_rows<<<GRID1((long long)N2 * DHID / 2), 0, stream>>>(hp1, hp1pk, (long long)N2 * DHID / 2);

    // ================= conv2: h2 = relu(mean@W2l + b2l + hp1@W2r) =================
    fill_f32<<<GRID1((long long)N2 * DHID), 0, stream>>>(bufA, 0.0f, (long long)N2 * DHID);
    fill_f32<<<GRID1(N2), 0, stream>>>(cnt, 0.0f, N2);
    scatter_add_feats<<<GRID1((long long)E_EDGES * (DHID / 4)), 0, stream>>>(hp1, src2, dst2, mask2, bufA, E_EDGES, DHID);
    scatter_count<<<GRID1(E_EDGES), 0, stream>>>(dst2, mask2, cnt, E_EDGES);
    mean_div_pack<<<GRID1((long long)N2 * DHID / 2), 0, stream>>>(bufA, cnt, meanpk, N2, DHID);
    gemm_dual_bias_relu<<<dim3(N2 / 16, DOUT / 64), dim3(32), 0, stream>>>(
        meanpk, hp1pk, w2lpk, w2rpk, b2l, h2, DHID, DOUT);

    // ================= pool2 (final; output goes straight to d_out) =================
    fill_f32<<<GRID1((long long)N2 * DOUT), 0, stream>>>(bufA, 0.0f, (long long)N2 * DOUT);
    scatter_add_feats<<<GRID1((long long)E_EDGES * (DOUT / 4)), 0, stream>>>(h2, src2, dst2, mask2, bufA, E_EDGES, DOUT);
    score_tanh<<<dim3(N2), dim3(128), 0, stream>>>(bufA, h2, Wp2_rel, bp2, Wp2_root, score, DOUT);
    topk_sort<<<dim3(B_GRAPHS), dim3(1024), 0, stream>>>(score, KEEP1, KEEP2, perm, vals);
    gather_scale<<<GRID1((long long)B_GRAPHS * KEEP2 * (DOUT / 4)), 0, stream>>>(
        h2, perm, vals, out, B_GRAPHS * KEEP2, DOUT);
    write_batch<<<GRID1(B_GRAPHS * KEEP2), 0, stream>>>(
        out + (size_t)B_GRAPHS * KEEP2 * DOUT, B_GRAPHS * KEEP2, KEEP2);
#undef GRID1
}
